// GATC_4904852652851
// MI455X (gfx1250) — matrix operand order repaired and checked by reading, not compiled, not run on hardware
//
#include <hip/hip_runtime.h>
#include <hip/hip_fp16.h>

typedef __attribute__((ext_vector_type(16))) _Float16 v16h;
typedef __attribute__((ext_vector_type(8)))  float    v8f;

#define N_NODES 50000
#define N_EDGES 1600000
#define N_ETOT  (N_EDGES + N_NODES)
#define N_G     64
#define FDIM    64
#define SLOPE   0.2f

// ---- float <-> order-preserving uint (for atomicMax on floats incl. negatives)
__device__ __forceinline__ unsigned f2ord(float f) {
  unsigned u = __float_as_uint(f);
  return (u & 0x80000000u) ? ~u : (u | 0x80000000u);
}
__device__ __forceinline__ float ord2f(unsigned u) {
  return (u & 0x80000000u) ? __uint_as_float(u & 0x7fffffffu) : __uint_as_float(~u);
}
#define ORD_NEG_INF 0x007FFFFFu   // f2ord(-inf)

// ---------------------------------------------------------------------------
// GEMM: H[N,64] = X[N,64] @ W[64,64], f16 WMMA with f32 accumulate.
// grid.x = N/16 row tiles; block = 128 threads = 4 waves; wave w -> cols [16w,16w+16)
// ---------------------------------------------------------------------------
__global__ __launch_bounds__(128)
void gatc_gemm64_wmma(const float* __restrict__ X, const float* __restrict__ W,
                      float* __restrict__ H) {
  const int wave = threadIdx.x >> 5;
  const int lane = threadIdx.x & 31;
  const int half = lane >> 4;
  const int ln   = lane & 15;
  const int row0 = blockIdx.x * 16;
  const int col0 = wave * 16;

  const float* __restrict__ xr = X + (size_t)(row0 + ln) * FDIM;

  v16h a0, a1, b0, b1;
#pragma unroll
  for (int v = 0; v < 8; ++v) {
    // A fragment (lane carries M, K = 16*half + 2v + pos, matching B)
    const int ka = 16 * half + 2 * v;
    float2 xa = *(const float2*)(xr + ka);          // K tile 0 (k = 0..31)
    a0[2 * v] = (_Float16)xa.x; a0[2 * v + 1] = (_Float16)xa.y;
    float2 xb = *(const float2*)(xr + 32 + ka);     // K tile 1 (k = 32..63)
    a1[2 * v] = (_Float16)xb.x; a1[2 * v + 1] = (_Float16)xb.y;

    // B fragment (B KxN, lane carries N, K = 16*half + 2v + pos)
    const int kb = 16 * half + 2 * v;
    b0[2 * v]     = (_Float16)W[(size_t)(kb)     * FDIM + col0 + ln];
    b0[2 * v + 1] = (_Float16)W[(size_t)(kb + 1) * FDIM + col0 + ln];
    b1[2 * v]     = (_Float16)W[(size_t)(kb + 32) * FDIM + col0 + ln];
    b1[2 * v + 1] = (_Float16)W[(size_t)(kb + 33) * FDIM + col0 + ln];
  }

  v8f c = {};
  c = __builtin_amdgcn_wmma_f32_16x16x32_f16(false, a0, false, b0, (short)0, c, false, false);
  c = __builtin_amdgcn_wmma_f32_16x16x32_f16(false, a1, false, b1, (short)0, c, false, false);

#pragma unroll
  for (int r = 0; r < 8; ++r) {
    const int m = r + 8 * half;
    H[(size_t)(row0 + m) * FDIM + col0 + ln] = c[r];
  }
}

// ---------------------------------------------------------------------------
// Per-node: attention dots s=h.a_src, d=h.a_dst; init m=-inf, z=0, agg=0
// ---------------------------------------------------------------------------
__global__ __launch_bounds__(256)
void gatc_nodeprep(const float* __restrict__ h, const float* __restrict__ a_s,
                   const float* __restrict__ a_d, float* __restrict__ s,
                   float* __restrict__ d, unsigned* __restrict__ m,
                   float* __restrict__ z, float* __restrict__ agg) {
  const int i = blockIdx.x * blockDim.x + threadIdx.x;
  if (i >= N_NODES) return;
  const float* hr = h + (size_t)i * FDIM;
  float ss = 0.f, dd = 0.f;
#pragma unroll
  for (int f = 0; f < FDIM; ++f) { const float v = hr[f]; ss += v * a_s[f]; dd += v * a_d[f]; }
  s[i] = ss; d[i] = dd;
  m[i] = ORD_NEG_INF;
  z[i] = 0.f;
  float4 z4 = {0.f, 0.f, 0.f, 0.f};
  float4* ar = (float4*)(agg + (size_t)i * FDIM);
#pragma unroll
  for (int f = 0; f < FDIM / 4; ++f) ar[f] = z4;
}

// ---------------------------------------------------------------------------
// Edge pass 1: e = LeakyReLU(s[src]+d[dst]); segment max into m[dst]
// ---------------------------------------------------------------------------
__global__ __launch_bounds__(256)
void gatc_edge_max(const int* __restrict__ ei, const float* __restrict__ s,
                   const float* __restrict__ d, float* __restrict__ ebuf,
                   unsigned* __restrict__ m) {
  const int e = blockIdx.x * blockDim.x + threadIdx.x;
  if (e >= N_ETOT) return;
  int src, dst;
  if (e < N_EDGES) { src = ei[e]; dst = ei[N_EDGES + e]; }
  else             { src = dst = e - N_EDGES; }
  float v = s[src] + d[dst];
  v = (v > 0.f) ? v : SLOPE * v;
  ebuf[e] = v;
  atomicMax(&m[dst], f2ord(v));
}

// ---------------------------------------------------------------------------
// Edge pass 2: p = exp(e - m[dst]); segment sum into z[dst]; ebuf := p
// ---------------------------------------------------------------------------
__global__ __launch_bounds__(256)
void gatc_edge_exp(const int* __restrict__ ei, float* __restrict__ ebuf,
                   const unsigned* __restrict__ m, float* __restrict__ z) {
  const int e = blockIdx.x * blockDim.x + threadIdx.x;
  if (e >= N_ETOT) return;
  const int dst = (e < N_EDGES) ? ei[N_EDGES + e] : (e - N_EDGES);
  const float p = __expf(ebuf[e] - ord2f(m[dst]));
  ebuf[e] = p;
  unsafeAtomicAdd(&z[dst], p);
}

// ---------------------------------------------------------------------------
// Edge pass 3: agg[dst] += (p/z[dst]) * h[src].  One wave32 per edge;
// each lane gathers 2 features (float2, hits L2) and does 2 hw f32 atomics.
// ---------------------------------------------------------------------------
__global__ __launch_bounds__(256)
void gatc_edge_scatter(const int* __restrict__ ei, const float* __restrict__ h,
                       const float* __restrict__ ebuf, const float* __restrict__ z,
                       float* __restrict__ agg) {
  const int eid = blockIdx.x * 8 + (threadIdx.x >> 5);
  if (eid >= N_ETOT) return;
  const int lane = threadIdx.x & 31;
  int src, dst;
  if (eid < N_EDGES) { src = ei[eid]; dst = ei[N_EDGES + eid]; }
  else               { src = dst = eid - N_EDGES; }
  const float alpha = ebuf[eid] / z[dst];
  const float2 hv = *(const float2*)(h + (size_t)src * FDIM + 2 * lane);
  float* ad = agg + (size_t)dst * FDIM + 2 * lane;
  unsafeAtomicAdd(ad,     alpha * hv.x);
  unsafeAtomicAdd(ad + 1, alpha * hv.y);
}

// ---------------------------------------------------------------------------
// Epilogue: out = relu(agg + b)
// ---------------------------------------------------------------------------
__global__ __launch_bounds__(256)
void gatc_epilogue(const float* __restrict__ agg, const float* __restrict__ b,
                   float* __restrict__ out) {
  const int idx = blockIdx.x * blockDim.x + threadIdx.x;
  if (idx >= N_NODES * FDIM) return;
  const float v = agg[idx] + b[idx & (FDIM - 1)];
  out[idx] = (v > 0.f) ? v : 0.f;
}

// ---------------------------------------------------------------------------
// Global max pool per graph. Post-ReLU values are >= 0, so raw uint atomicMax
// on the IEEE bits is order-correct with 0 init.
// ---------------------------------------------------------------------------
__global__ void gatc_pool_init(unsigned* __restrict__ gp) {
  const int i = blockIdx.x * blockDim.x + threadIdx.x;
  if (i < N_G * FDIM) gp[i] = 0u;
}

__global__ __launch_bounds__(256)
void gatc_pool(const float* __restrict__ h2, const int* __restrict__ batch,
               unsigned* __restrict__ gp) {
  const int idx = blockIdx.x * blockDim.x + threadIdx.x;
  if (idx >= N_NODES * FDIM) return;
  const int i = idx >> 6, f = idx & (FDIM - 1);
  atomicMax(&gp[batch[i] * FDIM + f], __float_as_uint(h2[idx]));
}

// ---------------------------------------------------------------------------
// Final classifier: out[g,c] = sum_f gp[g,f]*Wl[f,c] + bl[c]   (tiny)
// ---------------------------------------------------------------------------
__global__ void gatc_linear(const unsigned* __restrict__ gp, const float* __restrict__ Wl,
                            const float* __restrict__ bl, float* __restrict__ out) {
  const int idx = blockIdx.x * blockDim.x + threadIdx.x;
  if (idx >= N_G * 5) return;
  const int g = idx / 5, c = idx % 5;
  float acc = bl[c];
#pragma unroll 8
  for (int f = 0; f < FDIM; ++f) acc += __uint_as_float(gp[g * FDIM + f]) * Wl[f * 5 + c];
  out[idx] = acc;
}

// ---------------------------------------------------------------------------
extern "C" void kernel_launch(void* const* d_in, const int* in_sizes, int n_in,
                              void* d_out, int out_size, void* d_ws, size_t ws_size,
                              hipStream_t stream) {
  const float* x   = (const float*)d_in[0];
  const int*   ei  = (const int*)d_in[1];
  const int*   bat = (const int*)d_in[2];
  const float* W1  = (const float*)d_in[3];
  const float* a1s = (const float*)d_in[4];
  const float* a1d = (const float*)d_in[5];
  const float* b1  = (const float*)d_in[6];
  const float* W2  = (const float*)d_in[7];
  const float* a2s = (const float*)d_in[8];
  const float* a2d = (const float*)d_in[9];
  const float* b2  = (const float*)d_in[10];
  const float* Wl  = (const float*)d_in[11];
  const float* bl  = (const float*)d_in[12];

  // workspace carve-up (~46 MB)
  char* p = (char*)d_ws;
  float*    h    = (float*)p;    p += (size_t)N_NODES * FDIM * 4;
  float*    feat = (float*)p;    p += (size_t)N_NODES * FDIM * 4;
  float*    agg  = (float*)p;    p += (size_t)N_NODES * FDIM * 4;
  float*    sbuf = (float*)p;    p += (size_t)N_NODES * 4;
  float*    dbuf = (float*)p;    p += (size_t)N_NODES * 4;
  float*    zbuf = (float*)p;    p += (size_t)N_NODES * 4;
  unsigned* mbuf = (unsigned*)p; p += (size_t)N_NODES * 4;
  float*    ebuf = (float*)p;    p += (size_t)N_ETOT * 4;
  unsigned* gp   = (unsigned*)p; p += (size_t)N_G * FDIM * 4;

  const int nodeBlocks = (N_NODES + 255) / 256;
  const int edgeBlocks = (N_ETOT + 255) / 256;
  const int featBlocks = (N_NODES * FDIM + 255) / 256;
  const int scatBlocks = (N_ETOT + 7) / 8;

  auto layer = [&](const float* xin, const float* W, const float* as_, const float* ad_,
                   const float* b_, float* fout) {
    gatc_gemm64_wmma<<<N_NODES / 16, 128, 0, stream>>>(xin, W, h);
    gatc_nodeprep<<<nodeBlocks, 256, 0, stream>>>(h, as_, ad_, sbuf, dbuf, mbuf, zbuf, agg);
    gatc_edge_max<<<edgeBlocks, 256, 0, stream>>>(ei, sbuf, dbuf, ebuf, mbuf);
    gatc_edge_exp<<<edgeBlocks, 256, 0, stream>>>(ei, ebuf, mbuf, zbuf);
    gatc_edge_scatter<<<scatBlocks, 256, 0, stream>>>(ei, h, ebuf, zbuf, agg);
    gatc_epilogue<<<featBlocks, 256, 0, stream>>>(agg, b_, fout);
  };

  layer(x, W1, a1s, a1d, b1, feat);
  layer(feat, W2, a2s, a2d, b2, feat);   // stream-ordered: gemm reads feat before epilogue rewrites it

  gatc_pool_init<<<(N_G * FDIM + 255) / 256, 256, 0, stream>>>(gp);
  gatc_pool<<<featBlocks, 256, 0, stream>>>(feat, bat, gp);
  gatc_linear<<<(N_G * 5 + 255) / 256, 256, 0, stream>>>(gp, Wl, bl, (float*)d_out);
}
